// Head_64553358459001
// MI455X (gfx1250) — compile-verified, matrix operand order checked by reading, not run on hardware
//
#include <hip/hip_runtime.h>
#include <hip/hip_bf16.h>

typedef __attribute__((ext_vector_type(16))) _Float16 v16h;
typedef __attribute__((ext_vector_type(8)))  _Float16 v8h;
typedef __attribute__((ext_vector_type(8)))  float    v8f;
typedef __attribute__((ext_vector_type(4)))  float    v4f;
typedef __attribute__((ext_vector_type(4)))  int      v4i;

#define AS1 __attribute__((address_space(1)))
#define AS3 __attribute__((address_space(3)))

#define SEQ_LEN 8192
#define EMB     768
#define HD      64
#define INV_SCALE 0.125f   // 1/sqrt(64)

// LDS row strides (halves): padded for 16B alignment + bank-conflict avoidance
#define KSTR 72   // 32 keys x 64 dims, row stride 72  (144B, banks step 36)
#define VSTR 40   // 64 dims x 32 keys, row stride 40  (80B,  banks step 20)
#define PSTR 40   // 16 rows x 32 keys, row stride 40

#if __has_builtin(__builtin_amdgcn_global_load_async_to_lds_b128) && \
    __has_builtin(__builtin_amdgcn_s_wait_asynccnt)
#define USE_ASYNC_LDS 1
#else
#define USE_ASYNC_LDS 0
#endif

static __device__ __forceinline__ v8f wmma_f16(v16h a, v16h b, v8f c) {
    return __builtin_amdgcn_wmma_f32_16x16x32_f16(
        /*neg_a=*/false, a, /*neg_b=*/false, b,
        /*c_mod=*/(short)0, c, /*reuse_a=*/false, /*reuse_b=*/false);
}

// 16-bit 16x32 A/B fragment mapping (ISA 7.12.2):
// lane = (k/8 % 2)*16 + (m|n);  element e -> k = 8*hi + (e<8 ? e : e+8)
static __device__ __forceinline__ int kmap(int e, int hi) {
    return hi * 8 + (e < 8 ? e : e + 8);
}

// Load one 16-element f16 fragment slice for a 32-wide k-window starting at p
// (p = row base; must be 16B aligned).  Lowers to two 16B vector loads.
static __device__ __forceinline__ v16h ld_frag16(const _Float16* p, int hi) {
    const v8h a = *(const v8h*)(p + hi * 8);
    const v8h b = *(const v8h*)(p + 16 + hi * 8);
    return __builtin_shufflevector(a, b, 0, 1, 2, 3, 4, 5, 6, 7,
                                         8, 9, 10, 11, 12, 13, 14, 15);
}

// ---------------------------------------------------------------------------
// Kernel 0: one-time W transpose + f16 convert:
//   Wt[which][n][k] = (f16) W_which[k][n],  which in {Q,K,V}
// 288KB total; makes every qkv B-fragment read two contiguous b128 loads.
// ---------------------------------------------------------------------------
__global__ void __launch_bounds__(256) wt_prep(
        const float* __restrict__ Wq, const float* __restrict__ Wk,
        const float* __restrict__ Wv, _Float16* __restrict__ Wt) {
    const int idx = blockIdx.x * 256 + threadIdx.x;   // 3*64*768 total
    const int which = idx / (HD * EMB);
    const int rem   = idx - which * (HD * EMB);
    const int n     = rem / EMB;
    const int k     = rem - n * EMB;
    const float* W = (which == 0) ? Wq : (which == 1) ? Wk : Wv;
    Wt[idx] = (_Float16)W[(size_t)k * HD + n];
}

// ---------------------------------------------------------------------------
// Kernel 1: fused QKV projection.  One wave owns a 16-row strip of x and
// computes ALL of Q|K|V (12 accumulators) so x is streamed from HBM once.
// grid = SEQ/16 blocks of 32 threads; 12 WMMAs per 32-deep K step.
// ---------------------------------------------------------------------------
__global__ void __launch_bounds__(32) qkv_proj(
        const float* __restrict__ x, const _Float16* __restrict__ Wt,
        const float* __restrict__ bq, const float* __restrict__ bk,
        const float* __restrict__ bv,
        _Float16* __restrict__ Qh, _Float16* __restrict__ Kh,
        _Float16* __restrict__ Vh) {
    const int lane  = threadIdx.x & 31;
    const int lo    = lane & 15;
    const int hi    = lane >> 4;
    const int mbase = blockIdx.x * 16;

    const float* bs[3]   = {bq, bk, bv};
    _Float16*    dsts[3] = {Qh, Kh, Vh};

    const float* xrow = x + (size_t)(mbase + lo) * EMB;
    // Wt row for (which, g): n = which*64 + g*16 + lo, k-major (768 halves)
    const _Float16* wrow[3][4];
#pragma unroll
    for (int which = 0; which < 3; ++which)
#pragma unroll
        for (int g = 0; g < 4; ++g)
            wrow[which][g] = Wt + ((size_t)which * HD + g * 16 + lo) * EMB;

    v8f acc[3][4] = {};
    for (int k0 = 0; k0 < EMB; k0 += 32) {
        // A fragment: 4 x float4 contiguous loads, convert to f16
        const v4f x0 = *(const v4f*)(xrow + k0 + hi * 8);
        const v4f x1 = *(const v4f*)(xrow + k0 + hi * 8 + 4);
        const v4f x2 = *(const v4f*)(xrow + k0 + 16 + hi * 8);
        const v4f x3 = *(const v4f*)(xrow + k0 + 16 + hi * 8 + 4);
        v16h a;
#pragma unroll
        for (int s = 0; s < 4; ++s) {
            a[s]      = (_Float16)x0[s];
            a[4 + s]  = (_Float16)x1[s];
            a[8 + s]  = (_Float16)x2[s];
            a[12 + s] = (_Float16)x3[s];
        }
#pragma unroll
        for (int which = 0; which < 3; ++which) {
#pragma unroll
            for (int g = 0; g < 4; ++g) {
                const v16h bf = ld_frag16(wrow[which][g] + k0, hi);
                acc[which][g] = wmma_f16(a, bf, acc[which][g]);
            }
        }
    }

#pragma unroll
    for (int which = 0; which < 3; ++which) {
        const float scale = (which == 0) ? INV_SCALE : 1.0f;
#pragma unroll
        for (int g = 0; g < 4; ++g) {
            const float bias = bs[which][g * 16 + lo];
#pragma unroll
            for (int v = 0; v < 8; ++v) {
                const int row = mbase + v + 8 * hi;   // C layout: row = vgpr + 8*hi
                dsts[which][(size_t)row * HD + g * 16 + lo] =
                    (_Float16)((acc[which][g][v] + bias) * scale);
            }
        }
    }
}

// ---------------------------------------------------------------------------
// Cooperative stage of one 32-key chunk of K and V into LDS.
// K: async global->LDS b128 (CDNA5 path) into [key][dim] (stride KSTR).
// V: sync b128 load + transposed store into [dim][key] (stride VSTR) so the
//    P@V B-fragments read contiguous 16B groups from LDS.
// ---------------------------------------------------------------------------
static __device__ __forceinline__ void stage_chunk(
        const _Float16* __restrict__ Kh, const _Float16* __restrict__ Vh,
        int j0, _Float16* kdst, _Float16* vdstT, int tid) {
#pragma unroll
    for (int i = 0; i < 2; ++i) {                    // 2048 halves / (128 thr * 8)
        const int off = (i * 128 + tid) * 8;
        const int key = off >> 6;
        const int dim = off & 63;
        const _Float16* gsrc = Kh + (size_t)j0 * HD + off;
        _Float16*       ldst = kdst + key * KSTR + dim;
#if USE_ASYNC_LDS
        __builtin_amdgcn_global_load_async_to_lds_b128(
            (AS1 v4i*)(AS1 void*)gsrc,
            (AS3 v4i*)(AS3 void*)ldst,
            0, 0);
#else
        *(v8h*)ldst = *(const v8h*)gsrc;
#endif
    }
#pragma unroll
    for (int i = 0; i < 2; ++i) {
        const int off = (i * 128 + tid) * 8;
        const int key = off >> 6;
        const int dim = off & 63;
        const v8h val = *(const v8h*)(Vh + (size_t)j0 * HD + off);
#pragma unroll
        for (int s = 0; s < 8; ++s)
            vdstT[(dim + s) * VSTR + key] = val[s];
    }
}

// ---------------------------------------------------------------------------
// Kernel 2: causal flash attention.  128 threads = 4 waves; wave w owns
// query rows [qbase, qbase+16).  All waves run the block-max chunk count so
// barriers are uniform (fully-masked chunks are numerically inert), letting
// the block share double-buffered K/V LDS tiles loaded once per block.
// Per chunk: 4 WMMA (S = Q@K^T), online softmax via __shfl_xor in 16-lane
// halves, P transposed C->A layout through per-wave LDS (s_wait_dscnt),
// 4 WMMA (O += P@V).
// ---------------------------------------------------------------------------
__global__ void __launch_bounds__(128) flash_attn(
        const _Float16* __restrict__ Qh, const _Float16* __restrict__ Kh,
        const _Float16* __restrict__ Vh, float* __restrict__ out) {
    __shared__ _Float16 kbuf[2][32 * KSTR];
    __shared__ _Float16 vbufT[2][HD * VSTR];
    __shared__ _Float16 pbuf[4][16 * PSTR];

    const int tid   = threadIdx.x;
    const int lane  = tid & 31;
    const int w     = tid >> 5;
    const int lo    = lane & 15;
    const int hi    = lane >> 4;
    const int qbase = blockIdx.x * 64 + w * 16;

    // Q A-fragments (16 rows x 64 head-dim), pre-scaled by 1/sqrt(d)
    const _Float16* qrow = Qh + (size_t)(qbase + lo) * HD;
    const v16h aq0 = ld_frag16(qrow, hi);
    const v16h aq1 = ld_frag16(qrow + 32, hi);

    v8f o0 = {}, o1 = {}, o2 = {}, o3 = {};
    float m[8], l[8];
#pragma unroll
    for (int v = 0; v < 8; ++v) { m[v] = -__builtin_inff(); l[v] = 0.0f; }

    _Float16* pl = pbuf[w];
    const int nch = blockIdx.x * 2 + 2;              // uniform across the block

    stage_chunk(Kh, Vh, 0, kbuf[0], vbufT[0], tid);

    for (int it = 0; it < nch; ++it) {
        const int j0  = it * 32;
        const int cur = it & 1;

#if USE_ASYNC_LDS
        __builtin_amdgcn_s_wait_asynccnt(0);         // drain our K async loads
#endif
        __syncthreads();                             // chunk visible; prev readers done

        if (it + 1 < nch)
            stage_chunk(Kh, Vh, (it + 1) * 32, kbuf[cur ^ 1], vbufT[cur ^ 1], tid);

        const _Float16* kb = kbuf[cur];
        const _Float16* vt = vbufT[cur];

        // --- S = Q @ K^T (keys j0..j0+31) from LDS (ds_load_b128 pairs) ---
        const v16h bk0a = ld_frag16(kb + lo * KSTR, hi);
        const v16h bk0b = ld_frag16(kb + lo * KSTR + 32, hi);
        const v16h bk1a = ld_frag16(kb + (16 + lo) * KSTR, hi);
        const v16h bk1b = ld_frag16(kb + (16 + lo) * KSTR + 32, hi);

        v8f s0 = {}, s1 = {};
        s0 = wmma_f16(aq0, bk0a, s0);
        s0 = wmma_f16(aq1, bk0b, s0);
        s1 = wmma_f16(aq0, bk1a, s1);
        s1 = wmma_f16(aq1, bk1b, s1);

        // --- causal mask + online softmax --------------------------------
        float p0[8], p1[8], alpha[8];
#pragma unroll
        for (int v = 0; v < 8; ++v) {
            const int row = qbase + v + 8 * hi;
            const int c0 = j0 + lo, c1 = j0 + 16 + lo;
            float x0 = (c0 <= row) ? s0[v] : -__builtin_inff();
            float x1 = (c1 <= row) ? s1[v] : -__builtin_inff();
            float mx = fmaxf(x0, x1);
#pragma unroll
            for (int msk = 1; msk <= 8; msk <<= 1)
                mx = fmaxf(mx, __shfl_xor(mx, msk, 32));
            const float mnew = fmaxf(m[v], mx);      // finite after chunk 0
            const float e0 = __expf(x0 - mnew);      // exp(-inf) -> 0 masks
            const float e1 = __expf(x1 - mnew);
            float sum = e0 + e1;
#pragma unroll
            for (int msk = 1; msk <= 8; msk <<= 1)
                sum += __shfl_xor(sum, msk, 32);
            const float a = __expf(m[v] - mnew);
            m[v] = mnew;
            l[v] = l[v] * a + sum;
            alpha[v] = a;
            p0[v] = e0;
            p1[v] = e1;
        }
#pragma unroll
        for (int v = 0; v < 8; ++v) {
            o0[v] *= alpha[v]; o1[v] *= alpha[v];
            o2[v] *= alpha[v]; o3[v] *= alpha[v];
        }

        // --- transpose P (C layout) -> A fragment via per-wave LDS --------
#pragma unroll
        for (int v = 0; v < 8; ++v) {
            const int row = v + 8 * hi;
            pl[row * PSTR + lo]      = (_Float16)p0[v];
            pl[row * PSTR + 16 + lo] = (_Float16)p1[v];
        }
        asm volatile("s_wait_dscnt 0" ::: "memory"); // wave-local LDS fence
        const v16h pa = ld_frag16(pl + lo * PSTR, hi);

        // --- O += P @ V from transposed LDS (contiguous 16B groups) ------
        const v16h bv0 = ld_frag16(vt + (lo)      * VSTR, hi);
        const v16h bv1 = ld_frag16(vt + (16 + lo) * VSTR, hi);
        const v16h bv2 = ld_frag16(vt + (32 + lo) * VSTR, hi);
        const v16h bv3 = ld_frag16(vt + (48 + lo) * VSTR, hi);

        o0 = wmma_f16(pa, bv0, o0);
        o1 = wmma_f16(pa, bv1, o1);
        o2 = wmma_f16(pa, bv2, o2);
        o3 = wmma_f16(pa, bv3, o3);
    }

    // --- epilogue: O / l, fp32 output ------------------------------------
#pragma unroll
    for (int v = 0; v < 8; ++v) {
        const float inv = 1.0f / l[v];
        const size_t row = (size_t)(qbase + v + 8 * hi) * HD;
        out[row + lo]      = o0[v] * inv;
        out[row + 16 + lo] = o1[v] * inv;
        out[row + 32 + lo] = o2[v] * inv;
        out[row + 48 + lo] = o3[v] * inv;
    }
}

extern "C" void kernel_launch(void* const* d_in, const int* in_sizes, int n_in,
                              void* d_out, int out_size, void* d_ws, size_t ws_size,
                              hipStream_t stream) {
    const float* x  = (const float*)d_in[0];
    const float* Wq = (const float*)d_in[1];
    const float* bq = (const float*)d_in[2];
    const float* Wk = (const float*)d_in[3];
    const float* bk = (const float*)d_in[4];
    const float* Wv = (const float*)d_in[5];
    const float* bv = (const float*)d_in[6];

    _Float16* Qh = (_Float16*)d_ws;
    _Float16* Kh = Qh + (size_t)SEQ_LEN * HD;
    _Float16* Vh = Kh + (size_t)SEQ_LEN * HD;
    _Float16* Wt = Vh + (size_t)SEQ_LEN * HD;        // 3*64*768 halves

    wt_prep<<<dim3((3 * HD * EMB) / 256), 256, 0, stream>>>(Wq, Wk, Wv, Wt);

    qkv_proj<<<dim3(SEQ_LEN / 16), 32, 0, stream>>>(
        x, Wt, bq, bk, bv, Qh, Kh, Vh);

    flash_attn<<<dim3(SEQ_LEN / 64), 128, 0, stream>>>(
        Qh, Kh, Vh, (float*)d_out);
}
